// EnhancedPatchEmbedding_63806034149614
// MI455X (gfx1250) — compile-verified, matrix operand order checked by reading, not run on hardware
//
#include <hip/hip_runtime.h>
#include <hip/hip_bf16.h>
#include <stdint.h>

// ---------------- problem constants ----------------
#define IMG     224
#define PATCH   16
#define CIN     3
#define EDIM    768
#define GH      14              // IMG/PATCH
#define NPATCH  196             // GH*GH
#define BATCH   64
#define M_TOTAL (BATCH * NPATCH)   // 12544
#define K_TOTAL 3840               // PATCH*PATCH*CIN*5
#define LN_EPS  1e-5f

typedef __bf16 bf16;
typedef bf16  bf16x8  __attribute__((ext_vector_type(8)));
typedef bf16  bf16x16 __attribute__((ext_vector_type(16)));
typedef float f32x8   __attribute__((ext_vector_type(8)));

__device__ __forceinline__ bf16 f2bf(float f) {
  uint32_t u = __builtin_bit_cast(uint32_t, f);
  uint32_t r = u + 0x7FFFu + ((u >> 16) & 1u);   // round-to-nearest-even
  uint16_t h = (uint16_t)(r >> 16);
  return __builtin_bit_cast(bf16, h);
}

// CDNA5 async global->LDS copy (GVS mode: SGPR64 base + per-lane I32 byte offset).
// VDST = LDS byte address VGPR, tracked with ASYNCcnt.
__device__ __forceinline__ void async_ld16(uint32_t lds_addr, uint32_t voff, uint64_t sbase) {
  asm volatile("global_load_async_to_lds_b128 %0, %1, %2"
               :: "v"(lds_addr), "v"(voff), "s"(sbase) : "memory");
}
__device__ __forceinline__ void wait_async0() {
  asm volatile("s_wait_asynccnt 0x0" ::: "memory");
}

// ---------------------------------------------------------------------------
// Kernel 1: fused shift-stack + patch-unfold + fp32->bf16 (im2col).
// A[m][d], m = b*196 + gh*14 + gw, d = ph*240 + pw*15 + (s*3 + c)
// value = x[b, c, gh*16+ph - dx[s], gw*16+pw - dy[s]] (0 if OOB)
// ---------------------------------------------------------------------------
__global__ __launch_bounds__(256) void im2col_bf16(const float* __restrict__ x,
                                                   bf16* __restrict__ A) {
  const int d = blockIdx.x * 256 + threadIdx.x;   // 0..3839
  const int m = blockIdx.y;                       // 0..12543

  const int b  = m / NPATCH;  const int p  = m - b * NPATCH;
  const int gh = p / GH;      const int gw = p - gh * GH;
  const int ph = d / 240;     int r  = d - ph * 240;
  const int pw = r / 15;      const int c5 = r - pw * 15;
  const int s  = c5 / 3;      const int c  = c5 - s * 3;

  int dx = 0, dy = 0;
  switch (s) {                 // OFFSETS = [(0,4),(4,0),(0,-4),(-4,0)]
    case 1: dy =  4; break;
    case 2: dx =  4; break;
    case 3: dy = -4; break;
    case 4: dx = -4; break;
    default: break;
  }
  const int h = gh * PATCH + ph - dx;
  const int w = gw * PATCH + pw - dy;

  float v = 0.0f;
  if ((unsigned)h < IMG && (unsigned)w < IMG)
    v = x[(((size_t)b * CIN + c) * IMG + h) * IMG + w];

  A[(size_t)m * K_TOTAL + d] = f2bf(v);
}

// ---------------------------------------------------------------------------
// Kernel 2: weight fp32 -> bf16 (proj_w is (E, D), already N-major/K-contig)
// ---------------------------------------------------------------------------
__global__ __launch_bounds__(256) void weights_bf16(const float* __restrict__ w,
                                                    bf16* __restrict__ wb) {
  const size_t i = (size_t)blockIdx.x * 256 + threadIdx.x;
  wb[i] = f2bf(w[i]);
}

// ---------------------------------------------------------------------------
// Kernel 3: bf16 WMMA GEMM: H(M x E) = A(M x K) * Wb(E x K)^T, fp32 accum.
// Block tile 128x128, 8 waves (4 M-groups x 2 N-groups), wave tile 32x64
// = 8 accumulators of f32x8.  K step 32.  Double-buffered LDS tiles filled
// by CDNA5 async global->LDS b128 copies (ASYNCcnt).  Steady-state loop
// unconditionally prefetches slab kk+32; last slab is peeled (no branchy
// guard in the hot loop).
// ---------------------------------------------------------------------------
#define BM 128
#define BN 128
#define BK 32
#define LDSROW 40   // bf16 elements per row (80 bytes, 16B-aligned, conflict pad)

__global__ __launch_bounds__(256) void gemm_wmma_bf16(const bf16* __restrict__ A,
                                                      const bf16* __restrict__ Bw,
                                                      float* __restrict__ H) {
  __shared__ __align__(16) bf16 sA[2][BM * LDSROW];
  __shared__ __align__(16) bf16 sB[2][BN * LDSROW];

  const int tid   = threadIdx.x;
  const int lane  = tid & 31;
  const int wave  = tid >> 5;
  const int mg    = wave >> 1;          // 0..3
  const int ng    = wave & 1;           // 0..1
  const int row16 = lane & 15;
  const int khalf = lane >> 4;          // 0 or 1

  const int m0 = blockIdx.x * BM;
  const int n0 = blockIdx.y * BN;

  f32x8 acc[2][4];
  const f32x8 zero = {};
#pragma unroll
  for (int mi = 0; mi < 2; ++mi)
#pragma unroll
    for (int ni = 0; ni < 4; ++ni) acc[mi][ni] = zero;

  // --- async staging setup: each thread owns one 16B chunk in rows [0,64)
  //     and the matching chunk in rows [64,128) of both tiles ---
  const int ldr  = tid >> 2;            // 0..63
  const int kseg = (tid & 3) * 8;       // elem offset within 32-wide K slab

  const uint64_t baseA = (uint64_t)(uintptr_t)(A  + (size_t)m0 * K_TOTAL);
  const uint64_t baseB = (uint64_t)(uintptr_t)(Bw + (size_t)n0 * K_TOTAL);
  const uint32_t voff0   = (uint32_t)(((uint32_t)ldr * K_TOTAL + (uint32_t)kseg) * 2u);
  const uint32_t rowskip = (uint32_t)(64u * K_TOTAL * 2u);

  uint32_t ldsA0[2], ldsA1[2], ldsB0[2], ldsB1[2];
#pragma unroll
  for (int bufi = 0; bufi < 2; ++bufi) {
    ldsA0[bufi] = (uint32_t)(uintptr_t)&sA[bufi][ ldr       * LDSROW + kseg];
    ldsA1[bufi] = (uint32_t)(uintptr_t)&sA[bufi][(ldr + 64) * LDSROW + kseg];
    ldsB0[bufi] = (uint32_t)(uintptr_t)&sB[bufi][ ldr       * LDSROW + kseg];
    ldsB1[bufi] = (uint32_t)(uintptr_t)&sB[bufi][(ldr + 64) * LDSROW + kseg];
  }

  // stage one 32-wide K slab into LDS buffer `nb`
  auto stage = [&](int nb, int kk) {
    const uint32_t kb = voff0 + (uint32_t)(kk * 2);
    async_ld16(ldsA0[nb], kb,           baseA);
    async_ld16(ldsA1[nb], kb + rowskip, baseA);
    async_ld16(ldsB0[nb], kb,           baseB);
    async_ld16(ldsB1[nb], kb + rowskip, baseB);
  };

  // consume one staged slab: 6 fragment loads + 8 WMMAs
  // (CDNA5 16-bit A layout: elems 0..7 = K[8h..8h+7], elems 8..15 =
  //  K[16+8h..23+8h]; B mirrors with lane = N)
  auto compute = [&](int cb) {
    bf16x16 afrag[2], bfrag[4];
#pragma unroll
    for (int mi = 0; mi < 2; ++mi) {
      const bf16* p = &sA[cb][(mg * 32 + mi * 16 + row16) * LDSROW + khalf * 8];
      bf16x8 lo = *(const bf16x8*)p;
      bf16x8 hi = *(const bf16x8*)(p + 16);
      afrag[mi] = __builtin_shufflevector(lo, hi, 0,1,2,3,4,5,6,7,8,9,10,11,12,13,14,15);
    }
#pragma unroll
    for (int ni = 0; ni < 4; ++ni) {
      const bf16* p = &sB[cb][(ng * 64 + ni * 16 + row16) * LDSROW + khalf * 8];
      bf16x8 lo = *(const bf16x8*)p;
      bf16x8 hi = *(const bf16x8*)(p + 16);
      bfrag[ni] = __builtin_shufflevector(lo, hi, 0,1,2,3,4,5,6,7,8,9,10,11,12,13,14,15);
    }
#pragma unroll
    for (int mi = 0; mi < 2; ++mi)
#pragma unroll
      for (int ni = 0; ni < 4; ++ni)
        acc[mi][ni] = __builtin_amdgcn_wmma_f32_16x16x32_bf16(
            false, afrag[mi], false, bfrag[ni], (short)0, acc[mi][ni], false, false);
  };

  // prologue: stage slab 0 into buffer 0
  stage(0, 0);
  wait_async0();
  __syncthreads();

  // steady state: always prefetch the next slab (last slab peeled below)
  int buf = 0;
  for (int kk = 0; kk < K_TOTAL - BK; kk += BK) {
    stage(buf ^ 1, kk + BK);
    compute(buf);
    wait_async0();     // my prefetch landed in LDS
    __syncthreads();   // everyone's landed -> buffer published
    buf ^= 1;
  }
  compute(buf);        // final slab, nothing left to prefetch

  // epilogue: C layout — VGPR v, lanes 0..15: M=v, N=lane; lanes 16..31: M=v+8
#pragma unroll
  for (int mi = 0; mi < 2; ++mi) {
    const int mbase = m0 + mg * 32 + mi * 16 + 8 * khalf;
#pragma unroll
    for (int ni = 0; ni < 4; ++ni) {
      const int col = n0 + ng * 64 + ni * 16 + row16;
#pragma unroll
      for (int v = 0; v < 8; ++v)
        H[(size_t)(mbase + v) * EDIM + col] = acc[mi][ni][v];
    }
  }
}

// ---------------------------------------------------------------------------
// Kernel 4: in-place bias + LayerNorm over last dim (768) of H.
// One 256-thread block per row; each thread owns 3 elements.
// ---------------------------------------------------------------------------
__global__ __launch_bounds__(256) void bias_layernorm(float* __restrict__ H,
                                                      const float* __restrict__ bias,
                                                      const float* __restrict__ gamma,
                                                      const float* __restrict__ beta) {
  const int row = blockIdx.x;
  float* h = H + (size_t)row * EDIM;
  const int tid = threadIdx.x;

  const float v0 = h[tid]       + bias[tid];
  const float v1 = h[tid + 256] + bias[tid + 256];
  const float v2 = h[tid + 512] + bias[tid + 512];

  float s1 = v0 + v1 + v2;
  float s2 = v0 * v0 + v1 * v1 + v2 * v2;
#pragma unroll
  for (int off = 16; off > 0; off >>= 1) {
    s1 += __shfl_xor(s1, off, 32);
    s2 += __shfl_xor(s2, off, 32);
  }
  __shared__ float a1[8], a2[8];
  if ((tid & 31) == 0) { a1[tid >> 5] = s1; a2[tid >> 5] = s2; }
  __syncthreads();
  float S1 = 0.f, S2 = 0.f;
#pragma unroll
  for (int i = 0; i < 8; ++i) { S1 += a1[i]; S2 += a2[i]; }

  const float mu   = S1 * (1.0f / EDIM);
  const float var  = S2 * (1.0f / EDIM) - mu * mu;
  const float rstd = rsqrtf(var + LN_EPS);

  h[tid]       = (v0 - mu) * rstd * gamma[tid]       + beta[tid];
  h[tid + 256] = (v1 - mu) * rstd * gamma[tid + 256] + beta[tid + 256];
  h[tid + 512] = (v2 - mu) * rstd * gamma[tid + 512] + beta[tid + 512];
}

// ---------------------------------------------------------------------------
extern "C" void kernel_launch(void* const* d_in, const int* in_sizes, int n_in,
                              void* d_out, int out_size, void* d_ws, size_t ws_size,
                              hipStream_t stream) {
  const float* x     = (const float*)d_in[0];
  const float* pw    = (const float*)d_in[1];
  const float* pb    = (const float*)d_in[2];
  const float* gamma = (const float*)d_in[3];
  const float* beta  = (const float*)d_in[4];
  float* out = (float*)d_out;

  bf16* A  = (bf16*)d_ws;                                            // 96.3 MB
  bf16* Wb = (bf16*)((char*)d_ws + (size_t)M_TOTAL * K_TOTAL * 2);   // +5.9 MB

  im2col_bf16<<<dim3(K_TOTAL / 256, M_TOTAL), 256, 0, stream>>>(x, A);
  weights_bf16<<<(EDIM * K_TOTAL) / 256, 256, 0, stream>>>(pw, Wb);
  gemm_wmma_bf16<<<dim3(M_TOTAL / BM, EDIM / BN), 256, 0, stream>>>(A, Wb, out);
  bias_layernorm<<<M_TOTAL, 256, 0, stream>>>(out, pb, gamma, beta);
}
